// GlobalAttention_30855045054960
// MI455X (gfx1250) — matrix operand order repaired and checked by reading, not compiled, not run on hardware
//
#include <hip/hip_runtime.h>
#include <math.h>

#define Bsz   32
#define Lseq  2048
#define DSEQ  1024
#define DGLB  1024
#define Hn    16
#define DKd   64
#define DVd   64

#define SROW  1032   // padded K-stride (halves) for LDS tiles: staggers banks
#define SCR   68     // padded row stride (floats) for KS/VS scratch

typedef _Float16 half8 __attribute__((ext_vector_type(8)));
typedef _Float16 v16h  __attribute__((ext_vector_type(16)));
typedef float    v8f   __attribute__((ext_vector_type(8)));
typedef float    f4    __attribute__((ext_vector_type(4)));

__device__ __forceinline__ float gelu_exact(float x) {
    return 0.5f * x * (1.0f + erff(x * 0.70710678118654752f));
}

// ---------------------------------------------------------------------------
// Kernel 1: Q[b,h,k] = tanh( X[b,:] . Wq[h,:,k] )   -> workspace [B,H,64]
// ---------------------------------------------------------------------------
__global__ __launch_bounds__(256)
void qproj_kernel(const float* __restrict__ X,
                  const float* __restrict__ Wq,
                  float* __restrict__ Qout) {
    int gid = blockIdx.x * 256 + threadIdx.x;      // 0 .. 32767
    int b   = gid >> 10;                           // /1024
    int hk  = gid & 1023;                          // h*64 + k
    const float* x = X + (size_t)b * DGLB;
    const float* w = Wq + (size_t)(hk >> 6) * DGLB * DKd + (hk & 63);
    float acc = 0.f;
#pragma unroll 8
    for (int d = 0; d < DGLB; ++d)
        acc = fmaf(x[d], w[(size_t)d * DKd], acc);
    Qout[gid] = tanhf(acc);
}

// ---------------------------------------------------------------------------
// Kernel 2: fused KS/VS projection + online-softmax attention per (b,h)
// ---------------------------------------------------------------------------
__global__ __launch_bounds__(256)
void attn_kernel(const float* __restrict__ S,
                 const float* __restrict__ Wk,
                 const float* __restrict__ Wv,
                 const float* __restrict__ Q,
                 float* __restrict__ Y) {
    // f16 weights, transposed: ldsW[n][k], n = output column (0..63)
    __shared__ _Float16 ldsWk[64 * SROW];
    __shared__ _Float16 ldsWv[64 * SROW];
    // f16 S tile: 16 rows x 1024 (padded)
    __shared__ _Float16 ldsS[16 * SROW];
    // post-activation scratch tiles (f32)
    __shared__ float KSs[16 * SCR];
    __shared__ float VSs[16 * SCR];
    __shared__ float qv[64];
    __shared__ float pv[16];
    __shared__ float yacc[64];
    __shared__ float stM, stS, stScale;

    const int tid = threadIdx.x;
    const int b   = blockIdx.x >> 4;
    const int h   = blockIdx.x & 15;

    // ---- preload per-head weights, f32 -> f16, transposed into LDS ----
    {
        const float* wkg = Wk + (size_t)h * DSEQ * DKd;
        const float* wvg = Wv + (size_t)h * DSEQ * DVd;
        for (int i = tid; i < DSEQ * DKd / 4; i += 256) {
            int idx = i * 4;
            int k   = idx >> 6;   // row in [K][N] layout
            int n   = idx & 63;   // first of 4 consecutive columns
            f4 a = *(const f4*)(wkg + idx);
            f4 c = *(const f4*)(wvg + idx);
#pragma unroll
            for (int j = 0; j < 4; ++j) {
                ldsWk[(n + j) * SROW + k] = (_Float16)a[j];
                ldsWv[(n + j) * SROW + k] = (_Float16)c[j];
            }
        }
    }
    if (tid < 64) {
        qv[tid]   = Q[((size_t)(b * Hn) + h) * 64 + tid];
        yacc[tid] = 0.f;
    }
    if (tid == 0) { stM = -__builtin_inff(); stS = 0.f; stScale = 1.f; }

    // ---- per-wave WMMA tile assignment ----
    const int lane = tid & 31;
    const int w    = tid >> 5;     // wave 0..7
    const int proj = w >> 2;       // 0 = K-projection, 1 = V-projection
    const int nt   = w & 3;        // 16-col tile within the 64 output cols
    const int mlo  = lane & 15;
    const int hl   = lane >> 4;
    // A fragment base: lane holds row M=mlo, K run hl*16..+15
    const _Float16* arow = &ldsS[mlo * SROW + hl * 16];
    // B fragment base: lane holds column nt*16+mlo, K run hl*16..+15
    const _Float16* brow = (proj ? ldsWv : ldsWk)
                         + (size_t)(nt * 16 + mlo) * SROW + hl * 16;
    float* scr = proj ? VSs : KSs;

    const int sm   = tid >> 4;     // S-tile load: row
    const int sseg = tid & 15;     // S-tile load: 64-col segment
    const float* sbase = S + ((size_t)b * Lseq + sm) * DSEQ + sseg * 64;

    for (int lt = 0; lt < Lseq / 16; ++lt) {
        // ---- issue this tile's global reads BEFORE the barrier: their
        //      latency overlaps the barrier wait / previous y-update ----
        const float* srow = sbase + (size_t)lt * 16 * DSEQ;
        f4 xr[16];
#pragma unroll
        for (int j = 0; j < 16; ++j)
            xr[j] = *(const f4*)(srow + j * 4);

        __syncthreads();   // prev iteration's scratch/LDS consumers done

        // ---- convert f32 -> f16 and store the 16 x 1024 tile to LDS ----
        {
            _Float16* dst = &ldsS[sm * SROW + sseg * 64];
#pragma unroll
            for (int j = 0; j < 8; ++j) {
                f4 x0 = xr[2 * j];
                f4 x1 = xr[2 * j + 1];
                half8 hv;
                hv[0] = (_Float16)x0[0]; hv[1] = (_Float16)x0[1];
                hv[2] = (_Float16)x0[2]; hv[3] = (_Float16)x0[3];
                hv[4] = (_Float16)x1[0]; hv[5] = (_Float16)x1[1];
                hv[6] = (_Float16)x1[2]; hv[7] = (_Float16)x1[3];
                *(half8*)(dst + j * 8) = hv;
            }
        }
        __syncthreads();

        // ---- prefetch next tile into WGP$/L2 while WMMAs run ----
        if (lt + 1 < Lseq / 16) {
            const float* snext = srow + (size_t)16 * DSEQ;
            __builtin_prefetch(snext, 0, 3);        // global_prefetch_b8
            __builtin_prefetch(snext + 32, 0, 3);   // +128 B (2nd line)
        }

        // ---- 16x16 output tile over K=1024, two accumulators for ILP ----
        v8f c0 = {0.f,0.f,0.f,0.f,0.f,0.f,0.f,0.f};
        v8f c1 = {0.f,0.f,0.f,0.f,0.f,0.f,0.f,0.f};
#pragma unroll 2
        for (int kk = 0; kk < DSEQ; kk += 64) {
            half8 al0 = *(const half8*)(arow + kk);
            half8 ah0 = *(const half8*)(arow + kk + 8);
            half8 bl0 = *(const half8*)(brow + kk);
            half8 bh0 = *(const half8*)(brow + kk + 8);
            half8 al1 = *(const half8*)(arow + kk + 32);
            half8 ah1 = *(const half8*)(arow + kk + 40);
            half8 bl1 = *(const half8*)(brow + kk + 32);
            half8 bh1 = *(const half8*)(brow + kk + 40);
            v16h a0, b0, a1, b1;
#pragma unroll
            for (int j = 0; j < 8; ++j) {
                a0[j] = al0[j]; a0[8 + j] = ah0[j];
                b0[j] = bl0[j]; b0[8 + j] = bh0[j];
                a1[j] = al1[j]; a1[8 + j] = ah1[j];
                b1[j] = bl1[j]; b1[8 + j] = bh1[j];
            }
            c0 = __builtin_amdgcn_wmma_f32_16x16x32_f16(
                    false, a0, false, b0, (short)0, c0, false, false);
            c1 = __builtin_amdgcn_wmma_f32_16x16x32_f16(
                    false, a1, false, b1, (short)0, c1, false, false);
        }

        // ---- activation + spill tile to scratch ----
#pragma unroll
        for (int r = 0; r < 8; ++r) {
            float x = c0[r] + c1[r];
            float v = proj ? gelu_exact(x) : tanhf(x);
            scr[(r + 8 * hl) * SCR + nt * 16 + mlo] = v;
        }
        __syncthreads();

        // ---- logits + online softmax bookkeeping (lanes 0..15 of wave 0) --
        if (tid < 16) {
            const float* ks = &KSs[tid * SCR];
            float acc = 0.f;
#pragma unroll
            for (int k = 0; k < 64; ++k)
                acc = fmaf(qv[k], ks[k], acc);
            float logit = acc * 0.125f;            // 1/sqrt(64)
            float mx = logit;
#pragma unroll
            for (int off = 8; off > 0; off >>= 1)
                mx = fmaxf(mx, __shfl_xor(mx, off, 32));
            float prevM = stM;
            float newM  = fmaxf(prevM, mx);
            float pm    = __expf(logit - newM);
            float ps    = pm;
#pragma unroll
            for (int off = 8; off > 0; off >>= 1)
                ps += __shfl_xor(ps, off, 32);
            pv[tid] = pm;
            if (tid == 0) {
                float sc = __expf(prevM - newM);
                stScale = sc;
                stM     = newM;
                stS     = stS * sc + ps;
            }
        }
        __syncthreads();

        // ---- rescale + accumulate y[v] over this tile's 16 rows ----
        if (tid < 64) {
            float acc = yacc[tid] * stScale;
#pragma unroll
            for (int m = 0; m < 16; ++m)
                acc = fmaf(pv[m], VSs[m * SCR + tid], acc);
            yacc[tid] = acc;
        }
    }

    __syncthreads();
    if (tid < 64)
        Y[((size_t)b * Hn + h) * 64 + tid] = yacc[tid] / stS;
}

// ---------------------------------------------------------------------------
extern "C" void kernel_launch(void* const* d_in, const int* in_sizes, int n_in,
                              void* d_out, int out_size, void* d_ws, size_t ws_size,
                              hipStream_t stream) {
    (void)in_sizes; (void)n_in; (void)out_size; (void)ws_size;
    const float* X  = (const float*)d_in[0];
    const float* S  = (const float*)d_in[1];
    const float* Wq = (const float*)d_in[2];
    const float* Wk = (const float*)d_in[3];
    const float* Wv = (const float*)d_in[4];
    float* out = (float*)d_out;
    float* Qws = (float*)d_ws;    // [B,H,64] f32 = 128 KB

    qproj_kernel<<<(Bsz * Hn * DKd) / 256, 256, 0, stream>>>(X, Wq, Qws);
    attn_kernel<<<Bsz * Hn, 256, 0, stream>>>(S, Wk, Wv, Qws, out);
}
